// TreeRNN_11278584119712
// MI455X (gfx1250) — matrix operand order repaired and checked by reading, not compiled, not run on hardware
//
#include <hip/hip_runtime.h>

// Problem constants (reference: T, D, H, E, V)
#define TT 256
#define DD 512
#define HH 256
#define EE 256
#define VV 8192

#define NBLOCKS 64u   // persistent grid size (8 waves/block x 64 = 512 tiles)

typedef __attribute__((ext_vector_type(16))) __bf16 v16bf;
typedef __attribute__((ext_vector_type(8)))  __bf16 v8bf;
typedef __attribute__((ext_vector_type(8)))  float  v8f;

__device__ __forceinline__ v16bf cat8(v8bf lo, v8bf hi) {
    v16bf r;
#pragma unroll
    for (int i = 0; i < 8; ++i) { r[i] = lo[i]; r[8 + i] = hi[i]; }
    return r;
}

// A fragment: 16x32 bf16 tile, row-major source. `row` = this lane's row
// (M = lane&15). ISA layout: lanes 0-15 hold K=0..7 (V0-3) and K=16..23
// (V4-7); lanes 16-31 hold K=8..15 and K=24..31.
__device__ __forceinline__ v16bf load_a(const __bf16* row, int k0, int lane) {
    const __bf16* p = row + k0 + ((lane >> 4) << 3);
    return cat8(*(const v8bf*)p, *(const v8bf*)(p + 16));
}

// B fragment from LDS. Weight panels staged as [combo][16 rows][256 K] bf16
// (8KB per combo). B[k][n] = Wpanel[n_local][k]; lane n = lane&15, K-octets
// {8h, 16+8h} with h = lane>>4; two 16B ds_load_b128 per fragment.
__device__ __forceinline__ v16bf ld_b_lds(const __bf16* smem, int combo,
                                          int k0, int lane) {
    const __bf16* p = smem + combo * 4096 + (lane & 15) * 256 + k0 +
                      ((lane >> 4) << 3);
    return cat8(*(const v8bf*)p, *(const v8bf*)(p + 16));
}

__device__ __forceinline__ void mma(v8f& acc, v16bf a, v16bf b) {
    acc = __builtin_amdgcn_wmma_f32_16x16x32_bf16(
        /*neg_a=*/false, a, /*neg_b=*/false, b,
        /*c_mod=*/(short)0, acc, /*reuse_a=*/false, /*reuse_b=*/false);
}

__device__ __forceinline__ float sigm(float x) { return 1.f / (1.f + __expf(-x)); }

// 6-accumulator GEMM sweep over K=256: one A row stream (global, software
// pipelined) against 6 LDS weight panels (template combos -> immediate LDS
// offsets). bb[6] batch = 12 ds_load_b128 per chunk with staged dscnt waits.
// Unrolled x2 so the A-fragment pipeline rotation is pure register renaming.
template <int C0, int C1, int C2, int C3, int C4, int C5>
__device__ __forceinline__ void gemm6(const __bf16* smem, const __bf16* arow,
                                      int lane, v8f& a0, v8f& a1, v8f& a2,
                                      v8f& a3, v8f& a4, v8f& a5) {
    v16bf a = load_a(arow, 0, lane);
#pragma unroll 2
    for (int kc = 0; kc < 8; ++kc) {
        const int k0 = kc * 32;
        v16bf bb[6];
        bb[0] = ld_b_lds(smem, C0, k0, lane);
        bb[1] = ld_b_lds(smem, C1, k0, lane);
        bb[2] = ld_b_lds(smem, C2, k0, lane);
        bb[3] = ld_b_lds(smem, C3, k0, lane);
        bb[4] = ld_b_lds(smem, C4, k0, lane);
        bb[5] = ld_b_lds(smem, C5, k0, lane);
        v16bf an;
        if (kc < 7) an = load_a(arow, k0 + 32, lane);
        mma(a0, a, bb[0]);
        mma(a1, a, bb[1]);
        mma(a2, a, bb[2]);
        mma(a3, a, bb[3]);
        mma(a4, a, bb[4]);
        mma(a5, a, bb[5]);
        a = an;
    }
}

// Split-phase device barrier (monotone arrival counter in d_ws).
__device__ __forceinline__ void barrier_arrive(unsigned* bar) {
    __threadfence();          // make this step's h writes device-visible
    __syncthreads();
    if (threadIdx.x == 0)
        __hip_atomic_fetch_add(bar, 1u, __ATOMIC_ACQ_REL,
                               __HIP_MEMORY_SCOPE_AGENT);
}
__device__ __forceinline__ void barrier_wait(unsigned* bar, unsigned target) {
    if (threadIdx.x == 0) {
        while (__hip_atomic_load(bar, __ATOMIC_ACQUIRE,
                                 __HIP_MEMORY_SCOPE_AGENT) < target)
            __builtin_amdgcn_s_sleep(1);
    }
    __syncthreads();
}

// ---------------------------------------------------------------------------
// Prep kernels
// ---------------------------------------------------------------------------
__global__ void cvt_bf16_kernel(const float* __restrict__ src,
                                __bf16* __restrict__ dst, int n) {
    for (int i = blockIdx.x * blockDim.x + threadIdx.x; i < n;
         i += gridDim.x * blockDim.x)
        dst[i] = (__bf16)src[i];
}

__global__ void init_h_kernel(const float* __restrict__ h0,
                              float* __restrict__ h, __bf16* __restrict__ hbf,
                              unsigned* __restrict__ bar, int n) {
    if (blockIdx.x == 0 && threadIdx.x == 0) *bar = 0u;  // reset grid barrier
    for (int i = blockIdx.x * blockDim.x + threadIdx.x; i < n;
         i += gridDim.x * blockDim.x) {
        float v = h0[i];
        h[i] = v;
        hbf[i] = (__bf16)v;
    }
}

// ---------------------------------------------------------------------------
// Persistent dual-GRU recurrence with split-phase overlap.
//
// Block b owns output columns n0 = (b>>2)*16 for ALL 256 steps; its 12 weight
// panels (96 KB) are staged into LDS once. Waves cover m-tiles (b&3)*8+wave.
//
// Per step: Phase B (critical path) = h-side GEMM, 6 WMMAs/chunk accumulated
// into the x-side pre-activations computed LAST iteration; epilogue; arrive.
// Phase F (overlapped with other blocks' tails) = x-side GEMM for step t+1,
// which depends only on nodes/emb, executed between arrive and wait.
// ---------------------------------------------------------------------------
__global__ void __launch_bounds__(256, 1)
treernn_persistent(const int* __restrict__ nodes,
                   const unsigned char* __restrict__ edges,
                   const __bf16* __restrict__ emb_bf,
                   const __bf16* __restrict__ wci, const __bf16* __restrict__ wch,
                   const __bf16* __restrict__ wsi, const __bf16* __restrict__ wsh,
                   const float* __restrict__ cbi, const float* __restrict__ cbh,
                   const float* __restrict__ sbi, const float* __restrict__ sbh,
                   float* __restrict__ h0, float* __restrict__ h1,
                   __bf16* __restrict__ hbf0, __bf16* __restrict__ hbf1,
                   float* __restrict__ out, float* __restrict__ hfinal,
                   unsigned* __restrict__ bar) {
    extern __shared__ __bf16 smem[];   // 12 * 16 * 256 bf16 = 96 KB

    const int lane = threadIdx.x & 31;
    const int wave = threadIdx.x >> 5;
    const int b = blockIdx.x;
    const int n0 = (b >> 2) * 16;                   // column tile (fixed)
    const int m0 = ((b & 3) * 8 + wave) * 16;       // row tile per wave

    // ---- Stage the 12 weight panels for columns [n0, n0+16) into LDS ----
    // combo = wsel*3 + gate; wsel: 0=c_Wih 1=c_Whh 2=s_Wih 3=s_Whh
    {
        const __bf16* wsrc[4] = {wci, wch, wsi, wsh};
        for (int c = threadIdx.x; c < 12 * 512; c += 256) {
            const int combo = c >> 9;          // /512 chunks of 8 bf16
            const int rem = c & 511;
            const int row = rem >> 5;          // 0..15
            const int part = rem & 31;         // 32 x 8 bf16 = 256 K
            const int wsel = combo / 3;
            const int gate = combo - wsel * 3;
            const __bf16* src = wsrc[wsel] +
                (size_t)(gate * HH + n0 + row) * 256 + part * 8;
            *(v8bf*)(smem + combo * 4096 + row * 256 + part * 8) =
                *(const v8bf*)src;
        }
    }
    __syncthreads();

    // Biases for this lane's output column (constant across steps)
    const int col = n0 + (lane & 15);
    const float brc = cbi[col] + cbh[col];
    const float bzc = cbi[HH + col] + cbh[HH + col];
    const float bic = cbi[2 * HH + col];
    const float bnc = cbh[2 * HH + col];
    const float brs = sbi[col] + sbh[col];
    const float bzs = sbi[HH + col] + sbh[HH + col];
    const float bis = sbi[2 * HH + col];
    const float bns = sbh[2 * HH + col];

    const int d = m0 + (lane & 15);       // A-fragment row for this lane
    const int half = lane >> 4;           // C/D layout: VGPR i -> row i + 8*half

    const float*  hcur = h0;  float*  hnxt = h1;
    const __bf16* hbc = hbf0; __bf16* hbn = hbf1;

    // x-side gate pre-activations (persist across the barrier window)
    v8f xRc = {}, xZc = {}, xIc = {}, xRs = {}, xZs = {}, xIs = {};

    // Phase F for step 0 (x-side only; needs no h).
    gemm6<0, 1, 2, 6, 7, 8>(smem, emb_bf + (size_t)nodes[d] * EE, lane,
                            xRc, xZc, xIc, xRs, xZs, xIs);

    for (int t = 0; t < TT; ++t) {
        if (t) barrier_wait(bar, NBLOCKS * (unsigned)t);   // h(t) visible

        // Phase B: h-side GEMM on the critical path. r/z accumulate into the
        // x-side pre-acts; n-gate's hidden part stays separate.
        v8f aNc = {}, aNs = {};
        gemm6<3, 4, 5, 9, 10, 11>(smem, hbc + (size_t)d * HH, lane,
                                  xRc, xZc, aNc, xRs, xZs, aNs);

        float* out_t = out + (size_t)t * DD * HH;
#pragma unroll
        for (int i = 0; i < 8; ++i) {
            const int row = m0 + i + 8 * half;
            const float h_old = hcur[(size_t)row * HH + col];

            float r = sigm(xRc[i] + brc);
            float z = sigm(xZc[i] + bzc);
            float nn = tanhf(xIc[i] + bic + r * (aNc[i] + bnc));
            const float hc = (1.f - z) * nn + z * h_old;

            r = sigm(xRs[i] + brs);
            z = sigm(xZs[i] + bzs);
            nn = tanhf(xIs[i] + bis + r * (aNs[i] + bns));
            const float hs = (1.f - z) * nn + z * h_old;

            const float hv = edges[(size_t)t * DD + row] ? hc : hs;
            const size_t idx = (size_t)row * HH + col;
            out_t[idx] = hv;               // outs[t]
            hnxt[idx] = hv;                // f32 master state
            hbn[idx] = (__bf16)hv;         // bf16 shadow for next A-fragments
        }

        barrier_arrive(bar);               // signal h(t+1) written

        // Phase F for step t+1, overlapped with other blocks' phase B tails.
        if (t + 1 < TT) {
            const int node = nodes[(size_t)(t + 1) * DD + d];
            xRc = {}; xZc = {}; xIc = {}; xRs = {}; xZs = {}; xIs = {};
            gemm6<0, 1, 2, 6, 7, 8>(smem, emb_bf + (size_t)node * EE, lane,
                                    xRc, xZc, xIc, xRs, xZs, xIs);
        }

        float* tf = hnxt; hnxt = (float*)hcur; hcur = tf;
        __bf16* tb = hbn; hbn = (__bf16*)hbc; hbc = tb;
    }

    // Everyone must have written the final state before the strided copy.
    barrier_wait(bar, NBLOCKS * (unsigned)TT);
    for (int i = blockIdx.x * 256 + threadIdx.x; i < DD * HH; i += NBLOCKS * 256)
        hfinal[i] = hcur[i];
}

// ---------------------------------------------------------------------------
// Host launcher
// ---------------------------------------------------------------------------
extern "C" void kernel_launch(void* const* d_in, const int* in_sizes, int n_in,
                              void* d_out, int out_size, void* d_ws, size_t ws_size,
                              hipStream_t stream) {
    (void)in_sizes; (void)n_in; (void)out_size; (void)ws_size;

    const int* nodes           = (const int*)d_in[0];
    const unsigned char* edges = (const unsigned char*)d_in[1];
    const float* h_init        = (const float*)d_in[2];
    const float* emb           = (const float*)d_in[3];
    const float* cWih          = (const float*)d_in[4];
    const float* cWhh          = (const float*)d_in[5];
    const float* cbih          = (const float*)d_in[6];
    const float* cbhh          = (const float*)d_in[7];
    const float* sWih          = (const float*)d_in[8];
    const float* sWhh          = (const float*)d_in[9];
    const float* sbih          = (const float*)d_in[10];
    const float* sbhh          = (const float*)d_in[11];
    float* out                 = (float*)d_out;

    // Workspace carve-out (~7 MB), 256B-aligned slots.
    char* ws = (char*)d_ws;
    size_t off = 0;
    auto take = [&](size_t bytes) {
        size_t o = off;
        off = (off + bytes + 255) & ~(size_t)255;
        return o;
    };
    __bf16* emb_bf = (__bf16*)(ws + take((size_t)VV * EE * 2));
    __bf16* wci    = (__bf16*)(ws + take((size_t)3 * HH * EE * 2));
    __bf16* wch    = (__bf16*)(ws + take((size_t)3 * HH * HH * 2));
    __bf16* wsi    = (__bf16*)(ws + take((size_t)3 * HH * EE * 2));
    __bf16* wsh    = (__bf16*)(ws + take((size_t)3 * HH * HH * 2));
    float*  h0     = (float*)(ws + take((size_t)DD * HH * 4));
    float*  h1     = (float*)(ws + take((size_t)DD * HH * 4));
    __bf16* hbf0   = (__bf16*)(ws + take((size_t)DD * HH * 2));
    __bf16* hbf1   = (__bf16*)(ws + take((size_t)DD * HH * 2));
    unsigned* bar  = (unsigned*)(ws + take(256));

    // Per-call bf16 conversions + state/barrier init.
    cvt_bf16_kernel<<<512, 256, 0, stream>>>(emb, emb_bf, VV * EE);
    cvt_bf16_kernel<<<96, 256, 0, stream>>>(cWih, wci, 3 * HH * EE);
    cvt_bf16_kernel<<<96, 256, 0, stream>>>(cWhh, wch, 3 * HH * HH);
    cvt_bf16_kernel<<<96, 256, 0, stream>>>(sWih, wsi, 3 * HH * EE);
    cvt_bf16_kernel<<<96, 256, 0, stream>>>(sWhh, wsh, 3 * HH * HH);
    init_h_kernel<<<128, 256, 0, stream>>>(h_init, h0, hbf0, bar, DD * HH);

    // Single persistent kernel runs the entire 256-step recurrence.
    const size_t lds_bytes = (size_t)12 * 16 * 256 * sizeof(__bf16);  // 96 KB
    treernn_persistent<<<NBLOCKS, 256, lds_bytes, stream>>>(
        nodes, edges, emb_bf, wci, wch, wsi, wsh,
        cbih, cbhh, sbih, sbhh,
        h0, h1, hbf0, hbf1,
        out, out + (size_t)TT * DD * HH, bar);
}
